// Flow_32555852103694
// MI455X (gfx1250) — matrix operand order repaired and checked by reading, not compiled, not run on hardware
//
#include <hip/hip_runtime.h>
#include <hip/hip_bf16.h>

// ---------------- problem constants ----------------
#define NROWS    131072
#define DIMS_IN  16
#define DIMS_C   16
#define UNITS    512
#define BINS     10
#define FEAT_IN  24          // 8 cond dims + 16 c dims
#define FEAT_OUT 248         // 8*(3*BINS+1)
#define K0PAD    32          // FEAT_IN padded to one WMMA k-step
#define N2PAD    256         // FEAT_OUT padded to 16 col-tiles
#define BLOCKS   8

// ---------------- tiling ----------------
#define WAVES_PER_WG 4
#define ROWS_PER_WAVE 32     // two 16-row A tiles share each B fragment
#define ROWS_PER_WG  (WAVES_PER_WG * ROWS_PER_WAVE)   // 128

#define HSTRIDE 520                         // halfs per activation row (bank-spread)
#define HBUF_BYTES (ROWS_PER_WAVE * HSTRIDE * 2)      // 33280
#define PSTRIDE 260                         // f32 per params row: 32*260*4 == HBUF_BYTES
#define PANEL_STRIDE 520                    // halfs per B-panel column (bank-spread)
#define PANEL_BYTES (16 * PANEL_STRIDE * 2) // 16640
#define LDS_WAVE (2 * HBUF_BYTES)           // h0/h1 double buffer per wave
#define WG_LDS (WAVES_PER_WG * LDS_WAVE + 2 * PANEL_BYTES)  // 299520 <= 320KB

// workspace layout (bytes)
#define WS_W0T 0u
#define WS_W1T (WS_W0T + BLOCKS * UNITS * K0PAD * 2)
#define WS_W2T (WS_W1T + BLOCKS * UNITS * UNITS * 2)
#define WS_Z   (WS_W2T + BLOCKS * N2PAD * UNITS * 2)

typedef __attribute__((ext_vector_type(16))) _Float16 v16h;
typedef __attribute__((ext_vector_type(8)))  float    v8f;

union F16x16 { v16h v; _Float16 h[16]; };
union F32x8  { v8f  v; float    f[8];  };

// ---------------- CDNA5 async global->LDS copy (ASYNCcnt-tracked) ----------------
__device__ __forceinline__ unsigned lds_addr32(const void* p) {
    // low 32 bits of a generic pointer into LDS == LDS byte address
    return (unsigned)(unsigned long long)p;
}
__device__ __forceinline__ void async_copy_b128(const _Float16* g, _Float16* l) {
    asm volatile("global_load_async_to_lds_b128 %0, %1, off"
                 :: "v"(lds_addr32(l)), "v"(g) : "memory");
}
__device__ __forceinline__ void wait_async0() {
    asm volatile("s_wait_asynccnt 0x0" ::: "memory");
}

// stage one 16-col x 512-k f16 B panel (16 KB contiguous in global) into LDS,
// scattering to PANEL_STRIDE columns; 128 threads x 8 x b128
__device__ __forceinline__ void stage_panel(int tid, const _Float16* __restrict__ gsrc,
                                            _Float16* lpan) {
#pragma unroll
    for (int t = 0; t < 8; ++t) {
        const int cidx = tid + t * 128;      // 0..1023 16-byte chunks
        const int col  = cidx >> 6;          // 0..15
        const int off  = cidx & 63;          // 16B units within the column
        async_copy_b128(gsrc + col * 512 + off * 8,
                        lpan + col * PANEL_STRIDE + off * 8);
    }
}

// ---------------- prep: f16 transposed weights, z copy, out zero ----------------
__global__ __launch_bounds__(256) void flow_prep_kernel(
    const float* __restrict__ W0, const float* __restrict__ W1,
    const float* __restrict__ W2, const float* __restrict__ x,
    _Float16* __restrict__ w0t, _Float16* __restrict__ w1t,
    _Float16* __restrict__ w2t, float* __restrict__ z,
    float* __restrict__ out)
{
    const int i = blockIdx.x * blockDim.x + threadIdx.x;

    const int SZ0 = BLOCKS * UNITS * K0PAD;           // [b][n][k<=32]
    if (i < SZ0) {
        int b = i / (UNITS * K0PAD);
        int r = i % (UNITS * K0PAD);
        int n = r / K0PAD, k = r % K0PAD;
        float v = (k < FEAT_IN) ? W0[((size_t)b * FEAT_IN + k) * UNITS + n] : 0.f;
        w0t[i] = (_Float16)v;
    }
    const int SZ1 = BLOCKS * UNITS * UNITS;           // [b][n][k]
    if (i < SZ1) {
        int b = i / (UNITS * UNITS);
        int r = i % (UNITS * UNITS);
        int n = r / UNITS, k = r % UNITS;
        w1t[i] = (_Float16)W1[((size_t)b * UNITS + k) * UNITS + n];
    }
    const int SZ2 = BLOCKS * N2PAD * UNITS;           // [b][n<=256][k]
    if (i < SZ2) {
        int b = i / (N2PAD * UNITS);
        int r = i % (N2PAD * UNITS);
        int n = r / UNITS, k = r % UNITS;
        float v = (n < FEAT_OUT) ? W2[((size_t)b * UNITS + k) * FEAT_OUT + n] : 0.f;
        w2t[i] = (_Float16)v;
    }
    if (i < NROWS * DIMS_IN) z[i] = x[i];
    if (i < NROWS) out[i] = 0.f;
}

__device__ __forceinline__ float softplusf(float v) {
    return (v > 20.f) ? v : log1pf(expf(v));
}

// ---------------- fused coupling block: WMMA MLP + RQS ----------------
__global__ __launch_bounds__(128) void flow_block_kernel(
    float* __restrict__ z, const float* __restrict__ c,
    const float* __restrict__ b0, const float* __restrict__ b1,
    const float* __restrict__ b2,
    const _Float16* __restrict__ w0t, const _Float16* __restrict__ w1t,
    const _Float16* __restrict__ w2t,
    float* __restrict__ out, int blk)
{
    extern __shared__ char smem[];
    const int tid  = threadIdx.x;
    const int lane = tid & 31;
    const int wave = tid >> 5;
    const int rowbase = blockIdx.x * ROWS_PER_WG + wave * ROWS_PER_WAVE;

    _Float16* bufA = (_Float16*)(smem + wave * LDS_WAVE);               // h1
    _Float16* bufB = (_Float16*)(smem + wave * LDS_WAVE + HBUF_BYTES);  // h0 -> params
    float*    pbuf = (float*)bufB;
    _Float16* pan0 = (_Float16*)(smem + WAVES_PER_WG * LDS_WAVE);       // shared B panels
    _Float16* pan1 = pan0 + 16 * PANEL_STRIDE;

    // masks: m[blk][d] = ((d>>(blk/2))&1) ^ (blk&1); m==1 -> cond dim
    const int pshift = blk >> 1, par = blk & 1;
    unsigned condpack = 0u, transpack = 0u;
    int nc = 0, nt = 0;
    for (int d = 0; d < DIMS_IN; ++d) {
        int m = ((d >> pshift) & 1) ^ par;
        if (m) { condpack  |= (unsigned)d << (4 * nc); ++nc; }
        else   { transpack |= (unsigned)d << (4 * nt); ++nt; }
    }

    const _Float16* W0p = w0t + (size_t)blk * UNITS * K0PAD;
    const _Float16* W1p = w1t + (size_t)blk * UNITS * UNITS;
    const _Float16* W2p = w2t + (size_t)blk * N2PAD * UNITS;
    const float* b0p = b0 + blk * UNITS;
    const float* b1p = b1 + blk * UNITS;
    const float* b2p = b2 + blk * FEAT_OUT;

    const int r   = lane & 15;       // A row within tile
    const int kh  = lane >> 4;       // K-half select
    const int cn  = lane & 15;       // B/C column
    const int cr0 = (lane >> 4) * 8; // first C row in tile

    // kick off staging of W1 panel 0 so it overlaps layer-0 compute
    stage_panel(tid, W1p, pan0);

    // ---------- layer 0: A = [z[cond] | c], two 16-row tiles ----------
    F16x16 a0t0, a0t1;
#pragma unroll
    for (int i = 0; i < 8; ++i) {
        int f0 = kh * 8 + i;
        int f1 = 16 + kh * 8 + i;
#pragma unroll
        for (int ttile = 0; ttile < 2; ++ttile) {
            const long grow = rowbase + ttile * 16 + r;
            float v0 = 0.f, v1 = 0.f;
            if (f0 < 8)            v0 = z[grow * DIMS_IN + ((condpack >> (4 * f0)) & 15)];
            else if (f0 < FEAT_IN) v0 = c[grow * DIMS_C + (f0 - 8)];
            if (f1 < 8)            v1 = z[grow * DIMS_IN + ((condpack >> (4 * f1)) & 15)];
            else if (f1 < FEAT_IN) v1 = c[grow * DIMS_C + (f1 - 8)];
            F16x16& a = ttile ? a0t1 : a0t0;
            a.h[i]     = (_Float16)v0;
            a.h[8 + i] = (_Float16)v1;
        }
    }
    for (int n = 0; n < UNITS / 16; ++n) {
        F16x16 bf;
        const _Float16* bp = W0p + (size_t)(n * 16 + cn) * K0PAD + kh * 8;
#pragma unroll
        for (int i = 0; i < 16; ++i) bf.h[i] = bp[i + (i >> 3) * 8];
        v8f acc0 = {}, acc1 = {};
        acc0 = __builtin_amdgcn_wmma_f32_16x16x32_f16(false, a0t0.v, false, bf.v,
                                                      (short)0, acc0, false, false);
        acc1 = __builtin_amdgcn_wmma_f32_16x16x32_f16(false, a0t1.v, false, bf.v,
                                                      (short)0, acc1, false, false);
        float bias = b0p[n * 16 + cn];
        F32x8 c0; c0.v = acc0; F32x8 c1; c1.v = acc1;
#pragma unroll
        for (int i = 0; i < 8; ++i) {
            float v0 = c0.f[i] + bias, v1 = c1.f[i] + bias;
            bufB[(cr0 + i) * HSTRIDE + n * 16 + cn]        = (_Float16)(v0 > 0.f ? v0 : 0.f);
            bufB[(16 + cr0 + i) * HSTRIDE + n * 16 + cn]   = (_Float16)(v1 > 0.f ? v1 : 0.f);
        }
    }
    wait_async0();
    __syncthreads();                       // W1 panel 0 visible to all waves

    // ---------- layer 1: 512 -> 512, B panels staged async into LDS ----------
    for (int n = 0; n < UNITS / 16; ++n) {
        if (n + 1 < UNITS / 16)
            stage_panel(tid, W1p + (size_t)(n + 1) * 16 * UNITS, ((n + 1) & 1) ? pan1 : pan0);
        const _Float16* pb   = (n & 1) ? pan1 : pan0;
        const _Float16* bcol = pb + cn * PANEL_STRIDE + kh * 8;
        v8f acc0 = {}, acc1 = {};
#pragma unroll
        for (int k = 0; k < 16; ++k) {
            F16x16 bf;
#pragma unroll
            for (int i = 0; i < 16; ++i) bf.h[i] = bcol[k * 32 + i + (i >> 3) * 8];
            F16x16 a0, a1;
            const _Float16* pa0 = bufB + r * HSTRIDE + k * 32 + kh * 8;
            const _Float16* pa1 = pa0 + 16 * HSTRIDE;
#pragma unroll
            for (int i = 0; i < 8; ++i) {
                a0.h[i] = pa0[i]; a0.h[8 + i] = pa0[16 + i];
                a1.h[i] = pa1[i]; a1.h[8 + i] = pa1[16 + i];
            }
            acc0 = __builtin_amdgcn_wmma_f32_16x16x32_f16(false, a0.v, false, bf.v,
                                                          (short)0, acc0, false, false);
            acc1 = __builtin_amdgcn_wmma_f32_16x16x32_f16(false, a1.v, false, bf.v,
                                                          (short)0, acc1, false, false);
        }
        float bias = b1p[n * 16 + cn];
        F32x8 c0; c0.v = acc0; F32x8 c1; c1.v = acc1;
#pragma unroll
        for (int i = 0; i < 8; ++i) {
            float v0 = c0.f[i] + bias, v1 = c1.f[i] + bias;
            bufA[(cr0 + i) * HSTRIDE + n * 16 + cn]      = (_Float16)(v0 > 0.f ? v0 : 0.f);
            bufA[(16 + cr0 + i) * HSTRIDE + n * 16 + cn] = (_Float16)(v1 > 0.f ? v1 : 0.f);
        }
        wait_async0();
        __syncthreads();
    }

    // ---------- layer 2: 512 -> 248 (pad 256), params -> pbuf (f32) ----------
    stage_panel(tid, W2p, pan0);
    wait_async0();
    __syncthreads();
    for (int n = 0; n < N2PAD / 16; ++n) {
        if (n + 1 < N2PAD / 16)
            stage_panel(tid, W2p + (size_t)(n + 1) * 16 * UNITS, ((n + 1) & 1) ? pan1 : pan0);
        const _Float16* pb   = (n & 1) ? pan1 : pan0;
        const _Float16* bcol = pb + cn * PANEL_STRIDE + kh * 8;
        v8f acc0 = {}, acc1 = {};
#pragma unroll
        for (int k = 0; k < 16; ++k) {
            F16x16 bf;
#pragma unroll
            for (int i = 0; i < 16; ++i) bf.h[i] = bcol[k * 32 + i + (i >> 3) * 8];
            F16x16 a0, a1;
            const _Float16* pa0 = bufA + r * HSTRIDE + k * 32 + kh * 8;
            const _Float16* pa1 = pa0 + 16 * HSTRIDE;
#pragma unroll
            for (int i = 0; i < 8; ++i) {
                a0.h[i] = pa0[i]; a0.h[8 + i] = pa0[16 + i];
                a1.h[i] = pa1[i]; a1.h[8 + i] = pa1[16 + i];
            }
            acc0 = __builtin_amdgcn_wmma_f32_16x16x32_f16(false, a0.v, false, bf.v,
                                                          (short)0, acc0, false, false);
            acc1 = __builtin_amdgcn_wmma_f32_16x16x32_f16(false, a1.v, false, bf.v,
                                                          (short)0, acc1, false, false);
        }
        const int col = n * 16 + cn;
        float bias = (col < FEAT_OUT) ? b2p[col] : 0.f;
        F32x8 c0; c0.v = acc0; F32x8 c1; c1.v = acc1;
#pragma unroll
        for (int i = 0; i < 8; ++i) {
            pbuf[(cr0 + i) * PSTRIDE + col]      = c0.f[i] + bias;
            pbuf[(16 + cr0 + i) * PSTRIDE + col] = c1.f[i] + bias;
        }
        wait_async0();
        __syncthreads();
    }

    // ---------- RQS: lane <-> row, 8 trans dims per lane ----------
    {
        const long gr = rowbase + lane;
        const float* prow = pbuf + lane * PSTRIDE;
        float lsum = 0.f;
#pragma unroll
        for (int t = 0; t < 8; ++t) {
            const float* pp = prow + t * (3 * BINS + 1);
            float uw[BINS], uh[BINS], ud[BINS + 1];
#pragma unroll
            for (int k = 0; k < BINS; ++k) { uw[k] = pp[k]; uh[k] = pp[BINS + k]; }
#pragma unroll
            for (int k = 0; k < BINS + 1; ++k) ud[k] = pp[2 * BINS + k];

            float mw = uw[0], mh = uh[0];
#pragma unroll
            for (int k = 1; k < BINS; ++k) { mw = fmaxf(mw, uw[k]); mh = fmaxf(mh, uh[k]); }
            float sw = 0.f, sh = 0.f;
#pragma unroll
            for (int k = 0; k < BINS; ++k) {
                uw[k] = expf(uw[k] - mw); sw += uw[k];
                uh[k] = expf(uh[k] - mh); sh += uh[k];
            }
            const float fw = (1.f - 0.001f * BINS) / sw;
            const float fh = (1.f - 0.001f * BINS) / sh;
            float w[BINS], h[BINS], cw[BINS + 1], ch[BINS + 1], d[BINS + 1];
            cw[0] = 0.f; ch[0] = 0.f;
#pragma unroll
            for (int k = 0; k < BINS; ++k) {
                w[k] = 0.001f + fw * uw[k]; cw[k + 1] = cw[k] + w[k];
                h[k] = 0.001f + fh * uh[k]; ch[k + 1] = ch[k] + h[k];
            }
#pragma unroll
            for (int k = 0; k < BINS + 1; ++k) d[k] = 0.001f + softplusf(ud[k]);

            const int td = (transpack >> (4 * t)) & 15;
            const float xv = z[gr * DIMS_IN + td];

            int s = 0;
#pragma unroll
            for (int k = 0; k < BINS + 1; ++k) s += (xv >= cw[k]) ? 1 : 0;
            int idx = s - 1; idx = idx < 0 ? 0 : (idx > BINS - 1 ? BINS - 1 : idx);

            const float xk = cw[idx], wk = w[idx];
            const float yk = ch[idx], hk = h[idx];
            const float dk = d[idx],  dk1 = d[idx + 1];

            float th = (xv - xk) / wk; th = fminf(fmaxf(th, 0.f), 1.f);
            const float sl = hk / wk;
            const float t1 = th * (1.f - th);
            const float den = sl + (dk1 + dk - 2.f * sl) * t1;
            const float y = yk + hk * (sl * th * th + dk * t1) / den;
            const float omt = 1.f - th;
            const float ld = 2.f * logf(sl)
                           + logf(dk1 * th * th + 2.f * sl * t1 + dk * omt * omt)
                           - 2.f * logf(den);

            z[gr * DIMS_IN + td] = y;
            lsum += ld;
        }
        out[gr] = out[gr] + lsum;   // one lane per row -> no atomic needed
    }
}

// ---------------- launcher ----------------
extern "C" void kernel_launch(void* const* d_in, const int* in_sizes, int n_in,
                              void* d_out, int out_size, void* d_ws, size_t ws_size,
                              hipStream_t stream) {
    (void)in_sizes; (void)n_in; (void)out_size; (void)ws_size;
    const float* x  = (const float*)d_in[0];
    const float* c  = (const float*)d_in[1];
    const float* W0 = (const float*)d_in[2];
    const float* b0 = (const float*)d_in[3];
    const float* W1 = (const float*)d_in[4];
    const float* b1 = (const float*)d_in[5];
    const float* W2 = (const float*)d_in[6];
    const float* b2 = (const float*)d_in[7];
    float* out = (float*)d_out;

    char* ws = (char*)d_ws;
    _Float16* w0t = (_Float16*)(ws + WS_W0T);
    _Float16* w1t = (_Float16*)(ws + WS_W1T);
    _Float16* w2t = (_Float16*)(ws + WS_W2T);
    float*    z   = (float*)(ws + WS_Z);

    const int prep_n = NROWS * DIMS_IN;   // 2,097,152 >= every prep range
    flow_prep_kernel<<<(prep_n + 255) / 256, 256, 0, stream>>>(
        W0, W1, W2, x, w0t, w1t, w2t, z, out);

    const int grid = NROWS / ROWS_PER_WG; // 1024 WGs, 128 rows each
    for (int blk = 0; blk < BLOCKS; ++blk) {
        flow_block_kernel<<<grid, WAVES_PER_WG * 32, WG_LDS, stream>>>(
            z, c, b0, b1, b2, w0t, w1t, w2t, out, blk);
    }
}
